// GRUBaseline_88596585382812
// MI455X (gfx1250) — matrix operand order repaired and checked by reading, not compiled, not run on hardware
//
#include <hip/hip_runtime.h>

#define T_STEPS 365
#define NCELLS  65536
#define HID     128
#define WSTRIDE 144   // padded f16 row stride for w_hh in LDS (288B, 16B-aligned tiles)

typedef _Float16 v16h __attribute__((ext_vector_type(16)));
typedef _Float16 v8h  __attribute__((ext_vector_type(8)));
typedef float    v8f  __attribute__((ext_vector_type(8)));

union BU { unsigned u[8]; v16h h; };

static __device__ __forceinline__ unsigned pkf16(float a, float b) {
  union { _Float16 f[2]; unsigned w; } t;
  t.f[0] = (_Float16)a; t.f[1] = (_Float16)b;
  return t.w;
}
static __device__ __forceinline__ float unpk(unsigned w, int s) {
  union { unsigned w; _Float16 f[2]; } t; t.w = w;
  return (float)t.f[s];
}
static __device__ __forceinline__ float sigm(float x)   { return 1.0f / (1.0f + __expf(-x)); }
static __device__ __forceinline__ float tanh_f(float x) { float e = __expf(-2.0f*x); return (1.0f - e) / (1.0f + e); }

static __device__ __forceinline__ void ld8(const float* p, float* d) {
  float4 a = ((const float4*)p)[0];
  float4 b = ((const float4*)p)[1];
  d[0]=a.x; d[1]=a.y; d[2]=a.z; d[3]=a.w;
  d[4]=b.x; d[5]=b.y; d[6]=b.z; d[7]=b.w;
}

// Load one 16x32 f16 A-tile from LDS in WMMA A layout.
static __device__ __forceinline__ v16h loadA(const _Float16* lwhh, int row, int kb) {
  union { v16h h; struct { v8h lo, hi; } p; } r;
  const _Float16* ptr = lwhh + row * WSTRIDE + kb;
  r.p.lo = *(const v8h*)(ptr);
  r.p.hi = *(const v8h*)(ptr + 16);
  return r.h;
}

// One GRU timestep for 32 cells (two 16-cell column groups) of one wave.
// Reads hidden state from Brd*, writes next state into Bwr* (ping-pong).
static __device__ __forceinline__ void gru_step(
    int t,
    const float* __restrict__ precip, const float* __restrict__ temp,
    const _Float16* lwhh, const float* lw0, const float* lw1, const float* lb,
    const float* lbn, const float* lwo,
    BU* Brd0, BU* Bwr0, BU* Brd1, BU* Bwr1,
    int n, int half, int cell0, int cell1,
    float bo, float& smb0, float& smb1, float* __restrict__ out_fh)
{
  const float p0  = precip[(size_t)t * NCELLS + cell0];
  const float x20 = temp  [(size_t)t * NCELLS + cell0];
  const float p1  = precip[(size_t)t * NCELLS + cell1];
  const float x21 = temp  [(size_t)t * NCELLS + cell1];
  if (t + 1 < T_STEPS) {
    __builtin_prefetch(precip + (size_t)(t+1) * NCELLS + cell0, 0, 1);
    __builtin_prefetch(temp   + (size_t)(t+1) * NCELLS + cell0, 0, 1);
  }
  const float wt = (t == 0 || t == T_STEPS - 1) ? 0.5f : 1.0f;
  float ys0 = 0.0f, ys1 = 0.0f;

  for (int ct = 0; ct < 8; ++ct) {            // 16 hidden rows per iteration
    const int jr = 16*ct + 8*half;            // row base for this lane half

    float ta[8], tb[8], tc[8];
    v8f Cr0, Cr1, Cz0, Cz1, Cn0, Cn1;

    // r gate: C init = ir = x.w_ih + b
    ld8(lw0 + jr, ta); ld8(lw1 + jr, tb); ld8(lb + jr, tc);
    #pragma unroll
    for (int v = 0; v < 8; ++v) {
      Cr0[v] = fmaf(p0, ta[v], fmaf(x20, tb[v], tc[v]));
      Cr1[v] = fmaf(p1, ta[v], fmaf(x21, tb[v], tc[v]));
    }
    // z gate
    ld8(lw0 + 128 + jr, ta); ld8(lw1 + 128 + jr, tb); ld8(lb + 128 + jr, tc);
    #pragma unroll
    for (int v = 0; v < 8; ++v) {
      Cz0[v] = fmaf(p0, ta[v], fmaf(x20, tb[v], tc[v]));
      Cz1[v] = fmaf(p1, ta[v], fmaf(x21, tb[v], tc[v]));
    }
    // n gate: C accumulates hnew + b_n (inew added after the kt loop)
    ld8(lbn + jr, tc);
    #pragma unroll
    for (int v = 0; v < 8; ++v) { Cn0[v] = tc[v]; Cn1[v] = tc[v]; }

    // Each A-tile feeds both cell groups -> 2 WMMAs per ds_load_b128 pair.
    #pragma unroll
    for (int kt = 0; kt < 4; ++kt) {
      const int kb = 32*kt + 8*half;
      v16h Ar = loadA(lwhh,        16*ct + n, kb);
      Cr0 = __builtin_amdgcn_wmma_f32_16x16x32_f16(false, Ar, false, Brd0[kt].h, (short)0, Cr0, false, false);
      Cr1 = __builtin_amdgcn_wmma_f32_16x16x32_f16(false, Ar, false, Brd1[kt].h, (short)0, Cr1, false, false);
      v16h Az = loadA(lwhh, 128 +  16*ct + n, kb);
      Cz0 = __builtin_amdgcn_wmma_f32_16x16x32_f16(false, Az, false, Brd0[kt].h, (short)0, Cz0, false, false);
      Cz1 = __builtin_amdgcn_wmma_f32_16x16x32_f16(false, Az, false, Brd1[kt].h, (short)0, Cz1, false, false);
      v16h An = loadA(lwhh, 256 +  16*ct + n, kb);
      Cn0 = __builtin_amdgcn_wmma_f32_16x16x32_f16(false, An, false, Brd0[kt].h, (short)0, Cn0, false, false);
      Cn1 = __builtin_amdgcn_wmma_f32_16x16x32_f16(false, An, false, Brd1[kt].h, (short)0, Cn1, false, false);
    }

    // inew (n-gate x-path), loaded after the kt loop to shorten live ranges.
    float ign0[8], ign1[8], wo8[8];
    ld8(lw0 + 256 + jr, ta); ld8(lw1 + 256 + jr, tb); ld8(lb + 256 + jr, tc);
    #pragma unroll
    for (int v = 0; v < 8; ++v) {
      ign0[v] = fmaf(p0, ta[v], fmaf(x20, tb[v], tc[v]));
      ign1[v] = fmaf(p1, ta[v], fmaf(x21, tb[v], tc[v]));
    }
    ld8(lwo + jr, wo8);

    // K-permutation pi = swap bits 3<->4 of hidden index, applied by loadA's
    // split A reads (w_hh LDS is in natural order), makes C-layout row j land in this lane's own B slot:
    //   slot u = 4*(ct&1) + (v>>1), f16 half = v&1   (no cross-lane traffic)
    const int tB = ct >> 1;
    const int ub = 4 * (ct & 1);

    float hn0[8], hn1[8];
    #pragma unroll
    for (int v = 0; v < 8; ++v) {
      float rg0 = sigm(Cr0[v]);
      float zg0 = sigm(Cz0[v]);
      float nw0 = tanh_f(ign0[v] + rg0 * Cn0[v]);
      float ho0 = unpk(Brd0[tB].u[ub + (v >> 1)], v & 1);
      hn0[v] = nw0 + zg0 * (ho0 - nw0);
      ys0 = fmaf(hn0[v], wo8[v], ys0);

      float rg1 = sigm(Cr1[v]);
      float zg1 = sigm(Cz1[v]);
      float nw1 = tanh_f(ign1[v] + rg1 * Cn1[v]);
      float ho1 = unpk(Brd1[tB].u[ub + (v >> 1)], v & 1);
      hn1[v] = nw1 + zg1 * (ho1 - nw1);
      ys1 = fmaf(hn1[v], wo8[v], ys1);
    }
    #pragma unroll
    for (int u = 0; u < 4; ++u) {
      Bwr0[tB].u[ub + u] = pkf16(hn0[2*u], hn0[2*u + 1]);
      Bwr1[tB].u[ub + u] = pkf16(hn1[2*u], hn1[2*u + 1]);
    }

    if (t == T_STEPS - 1) {
      #pragma unroll
      for (int v = 0; v < 8; ++v) {
        out_fh[(size_t)cell0 * HID + jr + v] = hn0[v];
        out_fh[(size_t)cell1 * HID + jr + v] = hn1[v];
      }
    }
  }

  float y0 = ys0 + __shfl_xor(ys0, 16) + bo;  // combine half-wave partials
  float y1 = ys1 + __shfl_xor(ys1, 16) + bo;
  smb0 += wt * y0;
  smb1 += wt * y1;
}

extern "C" __global__ __launch_bounds__(128) void gru_wmma_kernel(
    const float* __restrict__ precip, const float* __restrict__ temp,
    const float* __restrict__ init_h, const float* __restrict__ w_ih,
    const float* __restrict__ w_hh,  const float* __restrict__ bvec,
    const float* __restrict__ b_n,   const float* __restrict__ w_out,
    const float* __restrict__ b_out, float* __restrict__ out)
{
  extern __shared__ __align__(16) unsigned char smem[];
  _Float16* lwhh = (_Float16*)smem;
  float* lw0 = (float*)(smem + 384 * WSTRIDE * 2);
  float* lw1 = lw0 + 384;
  float* lb  = lw1 + 384;
  float* lbn = lb  + 384;
  float* lwo = lbn + 128;

  const int tid = threadIdx.x;

  // Stage w_hh as f16 with K columns in natural order.
  for (int i = tid; i < 384 * HID; i += 128) {
    int r = i >> 7, k = i & 127;
    int kp = k;
    lwhh[r * WSTRIDE + kp] = (_Float16)w_hh[i];
  }
  for (int j = tid; j < 384; j += 128) {
    lw0[j] = w_ih[2*j]; lw1[j] = w_ih[2*j + 1]; lb[j] = bvec[j];
  }
  if (tid < 128) { lbn[tid] = b_n[tid]; lwo[tid] = w_out[tid]; }
  __syncthreads();

  const int lane  = tid & 31;
  const int wv    = tid >> 5;
  const int n     = lane & 15;   // cell column within tile
  const int half  = lane >> 4;   // half-wave
  const int cell0 = blockIdx.x * 128 + wv * 32 + n;
  const int cell1 = cell0 + 16;

  // Hidden state: two 16-cell groups x two ping-pong copies of 4 B-tiles.
  BU Ba0[4], Bb0[4], Ba1[4], Bb1[4];
  #pragma unroll
  for (int t = 0; t < 4; ++t)
    #pragma unroll
    for (int u = 0; u < 8; ++u) {
      // B slot (t, half, u) holds hidden rows j, j+1 with
      // j = 32t + 16*(u>=4) + 8*half + ((2u)&7)   (pi applied)
      int j = 32*t + ((u >= 4) ? 16 : 0) + 8*half + ((2*u) & 7);
      unsigned w = pkf16(init_h[j], init_h[j+1]);
      Ba0[t].u[u] = w; Ba1[t].u[u] = w;
      Bb0[t].u[u] = w; Bb1[t].u[u] = w;
    }

  float* out_smb = out;
  float* out_fh  = out + NCELLS;
  const float bo = b_out[0];
  float smb0 = 0.0f, smb1 = 0.0f;

  int t = 0;
  for (; t + 1 < T_STEPS; t += 2) {   // 182 pairs: steps 0..363
    gru_step(t,     precip, temp, lwhh, lw0, lw1, lb, lbn, lwo,
             Ba0, Bb0, Ba1, Bb1, n, half, cell0, cell1, bo, smb0, smb1, out_fh);
    gru_step(t + 1, precip, temp, lwhh, lw0, lw1, lb, lbn, lwo,
             Bb0, Ba0, Bb1, Ba1, n, half, cell0, cell1, bo, smb0, smb1, out_fh);
  }
  // final step t = 364 (T odd): state currently in Ba*
  gru_step(t, precip, temp, lwhh, lw0, lw1, lb, lbn, lwo,
           Ba0, Bb0, Ba1, Bb1, n, half, cell0, cell1, bo, smb0, smb1, out_fh);

  if (half == 0) {
    out_smb[cell0] = smb0;
    out_smb[cell1] = smb1;
  }
}

extern "C" void kernel_launch(void* const* d_in, const int* in_sizes, int n_in,
                              void* d_out, int out_size, void* d_ws, size_t ws_size,
                              hipStream_t stream) {
  const float* precip = (const float*)d_in[0];
  const float* temp   = (const float*)d_in[1];
  const float* init_h = (const float*)d_in[2];
  const float* w_ih   = (const float*)d_in[3];
  const float* w_hh   = (const float*)d_in[4];
  const float* bvec   = (const float*)d_in[5];
  const float* b_n    = (const float*)d_in[6];
  const float* w_out  = (const float*)d_in[7];
  const float* b_out  = (const float*)d_in[8];
  float* out = (float*)d_out;

  constexpr int SMEM = 384 * WSTRIDE * 2 + (384 * 3 + 128 + 128) * 4;  // ~113 KB
  gru_wmma_kernel<<<dim3(NCELLS / 128), dim3(128), SMEM, stream>>>(
      precip, temp, init_h, w_ih, w_hh, bvec, b_n, w_out, b_out, out);
}
